// Model_layer_56642028699718
// MI455X (gfx1250) — matrix operand order repaired and checked by reading, not compiled, not run on hardware
//
#include <hip/hip_runtime.h>
#include <cstddef>

// ---------------------------------------------------------------------------
// MI455X (gfx1250) implementation.
// Heavy GEMMs (GRU input/recurrent projections + MLPs) run on
// v_wmma_f32_16x16x32_bf16 (bf16 A/B, f32 accumulate). Gate nonlinearities and
// the recurrence are kept in f32. Weights are pre-packed once per call into
// WMMA B-fragment layout so every fragment is two coalesced b128 loads.
// Recurrent state h lives in LDS; one wave owns a 16-sequence tile for the
// whole T=64 scan (stage GRUs) or the whole 4096-step scan (eval GRU).
// Deterministic: no float atomics — per-tile partial sums + reduce-by-6.
// ---------------------------------------------------------------------------

typedef __attribute__((ext_vector_type(16))) __bf16 v16bf;
typedef __attribute__((ext_vector_type(8)))  __bf16 v8bf;
typedef __attribute__((ext_vector_type(8)))  float  v8f;

#define T_STEPS 64
#define BB      8
#define HID     128
#define EFFD    64
#define ATTD    32

__device__ __forceinline__ float sigm(float x) { return 1.0f / (1.0f + __expf(-x)); }

__device__ __forceinline__ v8f zero8() {
  v8f z;
#pragma unroll
  for (int i = 0; i < 8; i++) z[i] = 0.0f;
  return z;
}

__device__ __forceinline__ v8f wmma_b(v16bf a, v16bf b, v8f c) {
  // D = A(16x32 bf16) * B(32x16 bf16) + C(16x16 f32)
  return __builtin_amdgcn_wmma_f32_16x16x32_bf16(false, a, false, b, (short)0, c,
                                                 false, false);
}

// Pre-packed B fragment: 32 lanes x 16 bf16, contiguous per lane (32B/lane).
__device__ __forceinline__ v16bf bfrag(const __bf16* base, int fragIdx, int lane) {
  return *(const v16bf*)(base + (((size_t)fragIdx * 32 + (size_t)lane) << 4));
}

// A fragment (16x32 bf16) from global bf16, row-major, row stride ld.
// ISA layout: lane l holds row m=l&15; lanes<16: K = k0+[0..7], k0+16+[0..7];
// lanes>=16: K = k0+8+[0..7], k0+24+[0..7].
__device__ __forceinline__ v16bf a_frag_g(const __bf16* rowbase, int ld, int k0, int lane) {
  int row = lane & 15, hi = lane >> 4;
  const __bf16* p = rowbase + (size_t)row * ld;
  union { v16bf v; v8bf h[2]; } u;
  u.h[0] = *(const v8bf*)(p + k0 + hi * 8);
  u.h[1] = *(const v8bf*)(p + k0 + 16 + hi * 8);
  return u.v;
}

// A fragment gathered from an f32 matrix (LDS h-state), converted to bf16.
__device__ __forceinline__ v16bf a_frag_l(const float* h, int ld, int k0, int lane) {
  int row = lane & 15, hi = lane >> 4;
  const float* p = h + row * ld;
  int kb1 = k0 + hi * 8, kb2 = k0 + 16 + hi * 8;
  v16bf a;
#pragma unroll
  for (int e = 0; e < 8; e++) { a[e] = (__bf16)p[kb1 + e]; a[8 + e] = (__bf16)p[kb2 + e]; }
  return a;
}

// A fragment from f32 global with only BB valid rows (rest zero-padded).
__device__ __forceinline__ v16bf a_frag_seqf(const float* s, int ld, int k0, int lane) {
  int row = lane & 15, hi = lane >> 4;
  v16bf a;
  if (row < BB) {
    const float* p = s + row * ld;
    int kb1 = k0 + hi * 8, kb2 = k0 + 16 + hi * 8;
#pragma unroll
    for (int e = 0; e < 8; e++) { a[e] = (__bf16)p[kb1 + e]; a[8 + e] = (__bf16)p[kb2 + e]; }
  } else {
#pragma unroll
    for (int e = 0; e < 16; e++) a[e] = (__bf16)0.0f;
  }
  return a;
}

// ---------------------------------------------------------------------------
// Pack W [N,K] f32 (row n = output unit) into bf16 WMMA B fragments:
// out[((tn*(K/32)+tk)*32 + lane)*16 + e] = W[tn*16+(lane&15)][tk*32+((lane>>4)<<3)+(e&7)+((e>>3)<<4)]
// ---------------------------------------------------------------------------
__global__ void pack_b_kernel(const float* __restrict__ W, __bf16* __restrict__ o,
                              int N, int K) {
  int idx = blockIdx.x * blockDim.x + threadIdx.x;
  if (idx >= N * K) return;
  int e = idx & 15;
  int lane = (idx >> 4) & 31;
  int fragIdx = idx >> 9;
  int nk = K >> 5;
  int tn = fragIdx / nk, tk = fragIdx - tn * nk;
  int n = tn * 16 + (lane & 15);
  int k = tk * 32 + ((lane >> 4) << 3) + (e & 7) + ((e >> 3) << 4);
  o[idx] = (__bf16)W[(size_t)n * K + k];
}

// ---------------------------------------------------------------------------
// Interaction: x[t][n=r*B+b][d], d<32: M_r = rr . en_u ; 32<=d<96: rs . effect ;
// 96<=d<128: rs . en_l.  Output bf16 (feeds WMMA A-side directly).
// ---------------------------------------------------------------------------
__global__ void interaction_kernel(const float* __restrict__ rs,
                                   const float* __restrict__ rr,
                                   const float* __restrict__ enl,   // [B,L,T,ATT]
                                   const float* __restrict__ enu,   // [B,U,T,ATT]
                                   const float* __restrict__ eff,   // [L,T,B,EFF] or null
                                   __bf16* __restrict__ x, int R, int L, int U) {
  size_t idx = (size_t)blockIdx.x * blockDim.x + threadIdx.x;
  size_t total = (size_t)T_STEPS * R * BB * HID;
  if (idx >= total) return;
  int d = (int)(idx & 127);
  size_t tn = idx >> 7;
  size_t N = (size_t)R * BB;
  int t = (int)(tn / N);
  size_t n = tn - (size_t)t * N;
  int r = (int)(n / BB);
  int b = (int)(n - (size_t)r * BB);
  float acc = 0.0f;
  if (d < ATTD) {
    const float* rrp = rr + (((size_t)r * T_STEPS + t) * BB + b) * U;
    for (int u = 0; u < U; u++)
      acc += rrp[u] * enu[(((size_t)b * U + u) * T_STEPS + t) * ATTD + d];
  } else if (d < ATTD + EFFD) {
    if (eff) {
      int f = d - ATTD;
      const float* rsp = rs + (((size_t)r * T_STEPS + t) * BB + b) * L;
      for (int l = 0; l < L; l++)
        acc += rsp[l] * eff[(((size_t)l * T_STEPS + t) * BB + b) * EFFD + f];
    }
  } else {
    int a = d - (ATTD + EFFD);
    const float* rsp = rs + (((size_t)r * T_STEPS + t) * BB + b) * L;
    for (int l = 0; l < L; l++)
      acc += rsp[l] * enl[(((size_t)b * L + l) * T_STEPS + t) * ATTD + a];
  }
  x[idx] = (__bf16)acc;
}

// ---------------------------------------------------------------------------
// Relation GRU + fused MLP. One wave per 16-sequence tile (2 relations x B=8),
// full T=64 scan. h kept in LDS (f32). Partial output per tile (sum of its
// 2 relations) -> partial[tile][t][b][d]; reduce-by-6 gives effect[u][t][b][d].
// ---------------------------------------------------------------------------
struct RelParams {
  const __bf16 *ih1, *hh1, *w11, *w21;
  const __bf16 *ih3, *hh3, *w13, *w23;
  const float  *bih1, *bhh1, *b11, *b21;
  const float  *bih3, *bhh3, *b13, *b23;
};

__global__ __launch_bounds__(32)
void gru_relation_kernel(const __bf16* __restrict__ x, int N, RelParams P,
                         float* __restrict__ partial) {
  __shared__ float hls[16 * HID];
  __shared__ float mls[16 * HID];
  __shared__ float yls[16 * EFFD];
  int lane = threadIdx.x;
  int tile = blockIdx.x;
  int n0 = tile * 16;
  int r0 = n0 / BB;                        // first of 2 relations in this tile
  bool use3 = (r0 % 12) >= 8;              // types 1,2 -> params_r1 ; type 3 -> params_r3
  const __bf16* Wih = use3 ? P.ih3 : P.ih1;
  const __bf16* Whh = use3 ? P.hh3 : P.hh1;
  const __bf16* W1  = use3 ? P.w13 : P.w11;
  const __bf16* W2  = use3 ? P.w23 : P.w21;
  const float* bih = use3 ? P.bih3 : P.bih1;
  const float* bhh = use3 ? P.bhh3 : P.bhh1;
  const float* b1  = use3 ? P.b13 : P.b11;
  const float* b2  = use3 ? P.b23 : P.b21;

  for (int i = lane; i < 16 * HID; i += 32) hls[i] = 0.0f;
  __syncthreads();

  int colL = lane & 15;
  int rhalf = (lane >> 4) ? 8 : 0;

  for (int t = 0; t < T_STEPS; t++) {
    const __bf16* xrow = x + ((size_t)t * N + n0) * HID;
    v16bf ax[4], ah[4];
#pragma unroll
    for (int kt = 0; kt < 4; kt++) {
      ax[kt] = a_frag_g(xrow, HID, kt * 32, lane);
      ah[kt] = a_frag_l(hls, HID, kt * 32, lane);
    }
    __syncthreads();
#pragma unroll
    for (int j = 0; j < 8; j++) {        // 128 hidden cols, 16 at a time
      v8f ir = zero8(), iz = zero8(), inn = zero8();
      v8f hr = zero8(), hz = zero8(), hn = zero8();
#pragma unroll
      for (int kt = 0; kt < 4; kt++) {
        ir  = wmma_b(ax[kt], bfrag(Wih, (0  + j) * 4 + kt, lane), ir);
        iz  = wmma_b(ax[kt], bfrag(Wih, (8  + j) * 4 + kt, lane), iz);
        inn = wmma_b(ax[kt], bfrag(Wih, (16 + j) * 4 + kt, lane), inn);
        hr  = wmma_b(ah[kt], bfrag(Whh, (0  + j) * 4 + kt, lane), hr);
        hz  = wmma_b(ah[kt], bfrag(Whh, (8  + j) * 4 + kt, lane), hz);
        hn  = wmma_b(ah[kt], bfrag(Whh, (16 + j) * 4 + kt, lane), hn);
      }
      int c = j * 16 + colL;
      float bir = bih[c], biz = bih[HID + c], bin = bih[2 * HID + c];
      float bhr = bhh[c], bhz = bhh[HID + c], bhn = bhh[2 * HID + c];
#pragma unroll
      for (int i = 0; i < 8; i++) {
        int row = i + rhalf;
        float rg = sigm(ir[i] + bir + hr[i] + bhr);
        float zg = sigm(iz[i] + biz + hz[i] + bhz);
        float ng = tanhf(inn[i] + bin + rg * (hn[i] + bhn));
        float ho = hls[row * HID + c];
        hls[row * HID + c] = (1.0f - zg) * ng + zg * ho;
      }
    }
    __syncthreads();

    // Fused MLP: y = relu(h @ W1^T + b1) @ W2^T + b2
    v16bf am[4];
#pragma unroll
    for (int kt = 0; kt < 4; kt++) am[kt] = a_frag_l(hls, HID, kt * 32, lane);
#pragma unroll
    for (int j = 0; j < 8; j++) {
      v8f acc = zero8();
#pragma unroll
      for (int kt = 0; kt < 4; kt++) acc = wmma_b(am[kt], bfrag(W1, j * 4 + kt, lane), acc);
      int c = j * 16 + colL;
      float bv = b1[c];
#pragma unroll
      for (int i = 0; i < 8; i++) {
        float v = acc[i] + bv;
        mls[(i + rhalf) * HID + c] = v > 0.0f ? v : 0.0f;
      }
    }
    __syncthreads();
    v16bf az[4];
#pragma unroll
    for (int kt = 0; kt < 4; kt++) az[kt] = a_frag_l(mls, HID, kt * 32, lane);
#pragma unroll
    for (int j = 0; j < 4; j++) {        // 64 output dims
      v8f acc = zero8();
#pragma unroll
      for (int kt = 0; kt < 4; kt++) acc = wmma_b(az[kt], bfrag(W2, j * 4 + kt, lane), acc);
      int dcol = j * 16 + colL;
      float bv = b2[dcol];
#pragma unroll
      for (int i = 0; i < 8; i++)
        yls[(i + rhalf) * EFFD + dcol] = acc[i] + bv;
    }
    __syncthreads();
    // Deterministic in-tile sum of the 2 relations (rows b and b+8), same u.
    float* pp = partial + (((size_t)tile * T_STEPS + t) * BB) * EFFD;
    for (int i2 = lane; i2 < BB * EFFD; i2 += 32)
      pp[i2] = yls[i2] + yls[BB * EFFD + i2];
    __syncthreads();
  }
}

// effect[u][t][b][d] = sum_{k<6} partial[u*6+k][t][b][d]  (12 relations / 2 per tile)
__global__ void reduce6_kernel(const float* __restrict__ partial,
                               float* __restrict__ effect, int U) {
  size_t idx = (size_t)blockIdx.x * blockDim.x + threadIdx.x;
  size_t per_u = (size_t)T_STEPS * BB * EFFD;
  size_t total = (size_t)U * per_u;
  if (idx >= total) return;
  size_t u = idx / per_u, rem = idx - u * per_u;
  float s = 0.0f;
#pragma unroll
  for (int k = 0; k < 6; k++) s += partial[(u * 6 + k) * per_u + rem];
  effect[idx] = s;
}

// ---------------------------------------------------------------------------
// Eval GRU: 2 layers, 4096 sequential steps, batch 8 (padded to 16 rows).
// Single persistent wave; h0/h1 in LDS; final-hidden MLP -> out [2,8,64].
// ---------------------------------------------------------------------------
struct EvalParams {
  const __bf16 *ih0, *hh0, *ih1, *hh1, *w1, *w2;
  const float  *bih0, *bhh0, *bih1, *bhh1, *b1, *b2;
};

__global__ __launch_bounds__(32)
void eval_gru_kernel(const float* __restrict__ seq, EvalParams P,
                     float* __restrict__ out) {
  __shared__ float h0[16 * HID];
  __shared__ float h1[16 * HID];
  __shared__ float tm[16 * HID];
  int lane = threadIdx.x;
  for (int i = lane; i < 16 * HID; i += 32) { h0[i] = 0.0f; h1[i] = 0.0f; }
  __syncthreads();
  int colL = lane & 15;
  int rhalf = (lane >> 4) ? 8 : 0;

  for (int s = 0; s < 4096; s++) {
    const float* srow = seq + (size_t)s * BB * EFFD;
    // ---- layer 0 (input dim 64) ----
    v16bf ax0[2], ah0[4];
#pragma unroll
    for (int kt = 0; kt < 2; kt++) ax0[kt] = a_frag_seqf(srow, EFFD, kt * 32, lane);
#pragma unroll
    for (int kt = 0; kt < 4; kt++) ah0[kt] = a_frag_l(h0, HID, kt * 32, lane);
#pragma unroll
    for (int j = 0; j < 8; j++) {
      v8f ir = zero8(), iz = zero8(), inn = zero8();
      v8f hr = zero8(), hz = zero8(), hn = zero8();
#pragma unroll
      for (int kt = 0; kt < 2; kt++) {
        ir  = wmma_b(ax0[kt], bfrag(P.ih0, (0  + j) * 2 + kt, lane), ir);
        iz  = wmma_b(ax0[kt], bfrag(P.ih0, (8  + j) * 2 + kt, lane), iz);
        inn = wmma_b(ax0[kt], bfrag(P.ih0, (16 + j) * 2 + kt, lane), inn);
      }
#pragma unroll
      for (int kt = 0; kt < 4; kt++) {
        hr = wmma_b(ah0[kt], bfrag(P.hh0, (0  + j) * 4 + kt, lane), hr);
        hz = wmma_b(ah0[kt], bfrag(P.hh0, (8  + j) * 4 + kt, lane), hz);
        hn = wmma_b(ah0[kt], bfrag(P.hh0, (16 + j) * 4 + kt, lane), hn);
      }
      int c = j * 16 + colL;
      float bir = P.bih0[c], biz = P.bih0[HID + c], bin = P.bih0[2 * HID + c];
      float bhr = P.bhh0[c], bhz = P.bhh0[HID + c], bhn = P.bhh0[2 * HID + c];
#pragma unroll
      for (int i = 0; i < 8; i++) {
        int row = i + rhalf;
        float rg = sigm(ir[i] + bir + hr[i] + bhr);
        float zg = sigm(iz[i] + biz + hz[i] + bhz);
        float ng = tanhf(inn[i] + bin + rg * (hn[i] + bhn));
        float ho = h0[row * HID + c];
        h0[row * HID + c] = (1.0f - zg) * ng + zg * ho;
      }
    }
    __syncthreads();
    // ---- layer 1 (input = h0, dim 128) ----
    v16bf ax1[4], ah1[4];
#pragma unroll
    for (int kt = 0; kt < 4; kt++) {
      ax1[kt] = a_frag_l(h0, HID, kt * 32, lane);
      ah1[kt] = a_frag_l(h1, HID, kt * 32, lane);
    }
#pragma unroll
    for (int j = 0; j < 8; j++) {
      v8f ir = zero8(), iz = zero8(), inn = zero8();
      v8f hr = zero8(), hz = zero8(), hn = zero8();
#pragma unroll
      for (int kt = 0; kt < 4; kt++) {
        ir  = wmma_b(ax1[kt], bfrag(P.ih1, (0  + j) * 4 + kt, lane), ir);
        iz  = wmma_b(ax1[kt], bfrag(P.ih1, (8  + j) * 4 + kt, lane), iz);
        inn = wmma_b(ax1[kt], bfrag(P.ih1, (16 + j) * 4 + kt, lane), inn);
        hr  = wmma_b(ah1[kt], bfrag(P.hh1, (0  + j) * 4 + kt, lane), hr);
        hz  = wmma_b(ah1[kt], bfrag(P.hh1, (8  + j) * 4 + kt, lane), hz);
        hn  = wmma_b(ah1[kt], bfrag(P.hh1, (16 + j) * 4 + kt, lane), hn);
      }
      int c = j * 16 + colL;
      float bir = P.bih1[c], biz = P.bih1[HID + c], bin = P.bih1[2 * HID + c];
      float bhr = P.bhh1[c], bhz = P.bhh1[HID + c], bhn = P.bhh1[2 * HID + c];
#pragma unroll
      for (int i = 0; i < 8; i++) {
        int row = i + rhalf;
        float rg = sigm(ir[i] + bir + hr[i] + bhr);
        float zg = sigm(iz[i] + biz + hz[i] + bhz);
        float ng = tanhf(inn[i] + bin + rg * (hn[i] + bhn));
        float ho = h1[row * HID + c];
        h1[row * HID + c] = (1.0f - zg) * ng + zg * ho;
      }
    }
    __syncthreads();
  }

  // Final-hidden MLP for each layer: out[layer][b][dim]
  for (int li = 0; li < 2; li++) {
    const float* hsel = li ? h1 : h0;
    v16bf am[4];
#pragma unroll
    for (int kt = 0; kt < 4; kt++) am[kt] = a_frag_l(hsel, HID, kt * 32, lane);
#pragma unroll
    for (int j = 0; j < 8; j++) {
      v8f acc = zero8();
#pragma unroll
      for (int kt = 0; kt < 4; kt++) acc = wmma_b(am[kt], bfrag(P.w1, j * 4 + kt, lane), acc);
      int c = j * 16 + colL;
      float bv = P.b1[c];
#pragma unroll
      for (int i = 0; i < 8; i++) {
        float v = acc[i] + bv;
        tm[(i + rhalf) * HID + c] = v > 0.0f ? v : 0.0f;
      }
    }
    __syncthreads();
    v16bf az[4];
#pragma unroll
    for (int kt = 0; kt < 4; kt++) az[kt] = a_frag_l(tm, HID, kt * 32, lane);
#pragma unroll
    for (int j = 0; j < 4; j++) {
      v8f acc = zero8();
#pragma unroll
      for (int kt = 0; kt < 4; kt++) acc = wmma_b(az[kt], bfrag(P.w2, j * 4 + kt, lane), acc);
      int dcol = j * 16 + colL;
      float bv = P.b2[dcol];
#pragma unroll
      for (int i = 0; i < 8; i++) {
        int row = i + rhalf;
        if (row < BB) out[((size_t)li * BB + row) * EFFD + dcol] = acc[i] + bv;
      }
    }
    __syncthreads();
  }
}

// ---------------------------------------------------------------------------
// Host orchestration
// ---------------------------------------------------------------------------
extern "C" void kernel_launch(void* const* d_in, const int* in_sizes, int n_in,
                              void* d_out, int out_size, void* d_ws, size_t ws_size,
                              hipStream_t stream) {
  (void)n_in; (void)out_size; (void)ws_size;
  auto F = [&](int i) { return (const float*)d_in[i]; };

  const float *en1, *en2, *en3, *rs12, *rr12, *rs23, *rr23;
  const float *r1_wih, *r1_whh, *r1_bih, *r1_bhh, *r1_w1, *r1_b1, *r1_w2, *r1_b2;
  const float *r3_wih, *r3_whh, *r3_bih, *r3_bhh, *r3_w1, *r3_b1, *r3_w2, *r3_b2;
  const float *e0_wih, *e0_whh, *e0_bih, *e0_bhh;
  const float *e1_wih, *e1_whh, *e1_bih, *e1_bhh;
  const float *ev_w1, *ev_b1, *ev_w2, *ev_b2;

  if (in_sizes[3] == 128) {
    // Fully-sorted pytree leaf order (jax.tree_util.tree_leaves of the dict).
    en1 = F(0); en2 = F(1); en3 = F(2);
    ev_b1 = F(3); ev_b2 = F(4);
    e0_bhh = F(5); e0_bih = F(6); e0_whh = F(7); e0_wih = F(8);
    e1_bhh = F(9); e1_bih = F(10); e1_whh = F(11); e1_wih = F(12);
    ev_w1 = F(13); ev_w2 = F(14);
    r1_b1 = F(15); r1_b2 = F(16); r1_bhh = F(17); r1_bih = F(18);
    r1_whh = F(19); r1_wih = F(20); r1_w1 = F(21); r1_w2 = F(22);
    r3_b1 = F(23); r3_b2 = F(24); r3_bhh = F(25); r3_bih = F(26);
    r3_whh = F(27); r3_wih = F(28); r3_w1 = F(29); r3_w2 = F(30);
    // 31/32 = r_type_12/23 (structure is static; unused)
    rr12 = F(33); rr23 = F(34); rs12 = F(35); rs23 = F(36);
  } else {
    // Insertion order (dict order, nested dicts in insertion order).
    en1 = F(0); en2 = F(1); en3 = F(2);
    rs12 = F(3); rr12 = F(4); rs23 = F(5); rr23 = F(6);
    // 7/8 = r_type (unused)
    r1_wih = F(9); r1_whh = F(10); r1_bih = F(11); r1_bhh = F(12);
    r1_w1 = F(13); r1_b1 = F(14); r1_w2 = F(15); r1_b2 = F(16);
    r3_wih = F(17); r3_whh = F(18); r3_bih = F(19); r3_bhh = F(20);
    r3_w1 = F(21); r3_b1 = F(22); r3_w2 = F(23); r3_b2 = F(24);
    e0_wih = F(25); e0_whh = F(26); e0_bih = F(27); e0_bhh = F(28);
    e1_wih = F(29); e1_whh = F(30); e1_bih = F(31); e1_bhh = F(32);
    ev_w1 = F(33); ev_b1 = F(34); ev_w2 = F(35); ev_b2 = F(36);
  }

  // Workspace bump allocator (needs ~165 MB total).
  char* wp = (char*)d_ws;
  auto alloc = [&](size_t bytes) -> void* {
    void* p = (void*)wp;
    wp += (bytes + 255) & ~(size_t)255;
    return p;
  };
  __bf16* xbuf   = (__bf16*)alloc((size_t)T_STEPS * 6144 * HID * sizeof(__bf16)); // reused
  float* partial = (float*)alloc((size_t)384 * T_STEPS * BB * EFFD * sizeof(float)); // reused
  float* effect2 = (float*)alloc((size_t)32 * T_STEPS * BB * EFFD * sizeof(float));
  float* effect3 = (float*)alloc((size_t)64 * T_STEPS * BB * EFFD * sizeof(float));

  __bf16 *p_r1_ih = (__bf16*)alloc(49152 * 2), *p_r1_hh = (__bf16*)alloc(49152 * 2);
  __bf16 *p_r3_ih = (__bf16*)alloc(49152 * 2), *p_r3_hh = (__bf16*)alloc(49152 * 2);
  __bf16 *p_r1_w1 = (__bf16*)alloc(16384 * 2), *p_r1_w2 = (__bf16*)alloc(8192 * 2);
  __bf16 *p_r3_w1 = (__bf16*)alloc(16384 * 2), *p_r3_w2 = (__bf16*)alloc(8192 * 2);
  __bf16 *p_e0_ih = (__bf16*)alloc(24576 * 2), *p_e0_hh = (__bf16*)alloc(49152 * 2);
  __bf16 *p_e1_ih = (__bf16*)alloc(49152 * 2), *p_e1_hh = (__bf16*)alloc(49152 * 2);
  __bf16 *p_ev_w1 = (__bf16*)alloc(16384 * 2), *p_ev_w2 = (__bf16*)alloc(8192 * 2);

  auto pack = [&](const float* W, __bf16* o, int N, int K) {
    int tot = N * K;
    pack_b_kernel<<<(tot + 255) / 256, 256, 0, stream>>>(W, o, N, K);
  };
  pack(r1_wih, p_r1_ih, 384, 128); pack(r1_whh, p_r1_hh, 384, 128);
  pack(r3_wih, p_r3_ih, 384, 128); pack(r3_whh, p_r3_hh, 384, 128);
  pack(r1_w1, p_r1_w1, 128, 128);  pack(r1_w2, p_r1_w2, 64, 128);
  pack(r3_w1, p_r3_w1, 128, 128);  pack(r3_w2, p_r3_w2, 64, 128);
  pack(e0_wih, p_e0_ih, 384, 64);  pack(e0_whh, p_e0_hh, 384, 128);
  pack(e1_wih, p_e1_ih, 384, 128); pack(e1_whh, p_e1_hh, 384, 128);
  pack(ev_w1, p_ev_w1, 128, 128);  pack(ev_w2, p_ev_w2, 64, 128);

  RelParams RP;
  RP.ih1 = p_r1_ih; RP.hh1 = p_r1_hh; RP.w11 = p_r1_w1; RP.w21 = p_r1_w2;
  RP.ih3 = p_r3_ih; RP.hh3 = p_r3_hh; RP.w13 = p_r3_w1; RP.w23 = p_r3_w2;
  RP.bih1 = r1_bih; RP.bhh1 = r1_bhh; RP.b11 = r1_b1; RP.b21 = r1_b2;
  RP.bih3 = r3_bih; RP.bhh3 = r3_bhh; RP.b13 = r3_b1; RP.b23 = r3_b2;

  // ---- Stage 1: 384 relations, N = 3072 sequences ----
  {
    size_t tot = (size_t)T_STEPS * 384 * BB * HID;
    interaction_kernel<<<(unsigned)((tot + 255) / 256), 256, 0, stream>>>(
        rs12, rr12, en1, en2, nullptr, xbuf, 384, 16, 32);
  }
  gru_relation_kernel<<<3072 / 16, 32, 0, stream>>>(xbuf, 3072, RP, partial);
  {
    size_t tot = (size_t)32 * T_STEPS * BB * EFFD;
    reduce6_kernel<<<(unsigned)((tot + 255) / 256), 256, 0, stream>>>(partial, effect2, 32);
  }

  // ---- Stage 2: 768 relations, N = 6144 sequences ----
  {
    size_t tot = (size_t)T_STEPS * 768 * BB * HID;
    interaction_kernel<<<(unsigned)((tot + 255) / 256), 256, 0, stream>>>(
        rs23, rr23, en2, en3, effect2, xbuf, 768, 32, 64);
  }
  gru_relation_kernel<<<6144 / 16, 32, 0, stream>>>(xbuf, 6144, RP, partial);
  {
    size_t tot = (size_t)64 * T_STEPS * BB * EFFD;
    reduce6_kernel<<<(unsigned)((tot + 255) / 256), 256, 0, stream>>>(partial, effect3, 64);
  }

  // ---- Eval GRU over 4096-step sequence, batch 8 ----
  EvalParams EP;
  EP.ih0 = p_e0_ih; EP.hh0 = p_e0_hh; EP.ih1 = p_e1_ih; EP.hh1 = p_e1_hh;
  EP.w1 = p_ev_w1; EP.w2 = p_ev_w2;
  EP.bih0 = e0_bih; EP.bhh0 = e0_bhh; EP.bih1 = e1_bih; EP.bhh1 = e1_bhh;
  EP.b1 = ev_b1; EP.b2 = ev_b2;
  eval_gru_kernel<<<1, 32, 0, stream>>>(effect3, EP, (float*)d_out);
}
